// ModifiedTripletMarginLoss_77790447665533
// MI455X (gfx1250) — compile-verified, matrix operand order checked by reading, not run on hardware
//
#include <hip/hip_runtime.h>
#include <hip/hip_bf16.h>
#include <math.h>

typedef _Float16 v8h  __attribute__((ext_vector_type(8)));
typedef _Float16 v16h __attribute__((ext_vector_type(16)));
typedef float    v8f  __attribute__((ext_vector_type(8)));
typedef int      v4i  __attribute__((ext_vector_type(4)));

#if defined(__has_builtin)
#if __has_builtin(__builtin_amdgcn_global_load_async_to_lds_b128)
#define USE_ASYNC_LDS 1
#endif
#endif
#ifndef USE_ASYNC_LDS
#define USE_ASYNC_LDS 0
#endif

namespace {
constexpr int kC = 256;    // feature channels
constexpr int kK = 20;     // classes
constexpr int kP = 2048;   // positives per class
constexpr int kQ = 2048;   // negatives per class
constexpr float kMargin = 0.3f;
constexpr float kEps = 0.0001f / 256.0f;

// LDS B-tile: 16 rows x 256 f16, rows padded to 528B (33*16B) to spread the
// 16-lane b128 fragment reads across banks (512B stride would hit one bank).
constexpr int kLdsRowB   = 528;
constexpr int kLdsTileB  = 16 * kLdsRowB;   // 8448 B per buffer

__device__ __forceinline__ float sigmoidf_fast(float x) {
  return 1.0f / (1.0f + __expf(-x));
}

// ---- fragment loads (CDNA5 wave32 WMMA A/B register layout) ----
//   lane l<16  : row = l,    K = {kk*32+0..7}  in halves 0-3, {+16..23} in 4-7
//   lane l>=16 : row = l-16, K = {kk*32+8..15} in halves 0-3, {+24..31} in 4-7
__device__ __forceinline__ v16h load_frag_global(const _Float16* __restrict__ base,
                                                 int row0, int kk, int lane) {
  int r  = lane & 15;
  int hi = (lane >> 4) & 1;
  const _Float16* p = base + (size_t)(row0 + r) * kC + kk * 32 + hi * 8;
  v8h lo  = *(const v8h*)(p);
  v8h hi8 = *(const v8h*)(p + 16);
  v16h out;
  #pragma unroll
  for (int i = 0; i < 8; ++i) { out[i] = lo[i]; out[i + 8] = hi8[i]; }
  return out;
}

__device__ __forceinline__ v16h load_frag_lds(const char* tile, int kk, int lane) {
  int r  = lane & 15;
  int hi = (lane >> 4) & 1;
  const char* p = tile + r * kLdsRowB + kk * 64 + hi * 16;
  v8h lo  = *(const v8h*)(p);        // ds_load_b128
  v8h hi8 = *(const v8h*)(p + 32);
  v16h out;
  #pragma unroll
  for (int i = 0; i < 8; ++i) { out[i] = lo[i]; out[i + 8] = hi8[i]; }
  return out;
}

// ---- cooperative B-tile staging: global (16x256 f16, contiguous) -> LDS ----
// 512 chunks of 16B; 128 threads x 4 chunks each.
__device__ __forceinline__ void stage_tile(const _Float16* __restrict__ gsrc,
                                           char* lds_dst, int tid) {
  #pragma unroll
  for (int i = 0; i < 4; ++i) {
    int c   = tid + 128 * i;        // 16B chunk id 0..511
    int row = c >> 5;
    int col = c & 31;
    const _Float16* g = gsrc + (size_t)row * kC + col * 8;
    char* l = lds_dst + row * kLdsRowB + col * 16;
#if USE_ASYNC_LDS
    // GLOBAL_LOAD_ASYNC_TO_LDS_B128: memory -> LDS, tracked by ASYNCcnt.
    __builtin_amdgcn_global_load_async_to_lds_b128(
        (__attribute__((address_space(1))) v4i*)const_cast<_Float16*>(g),
        (__attribute__((address_space(3))) v4i*)l, 0, 0);
#else
    v8h v = *(const v8h*)g;
    *(v8h*)l = v;                   // ds_store_b128
#endif
  }
}

__device__ __forceinline__ void wait_stage() {
#if USE_ASYNC_LDS
#if __has_builtin(__builtin_amdgcn_s_wait_asynccnt)
  __builtin_amdgcn_s_wait_asynccnt(0);
#else
  asm volatile("s_wait_asynccnt 0x0" ::: "memory");
#endif
#endif
  __syncthreads();
}
} // namespace

// ---------------------------------------------------------------------------
// Kernel A: gather rows via idx, apply sigmoid, emit f16 copy + f32 row norm.
// One wave32 per row, 8 waves per block.
// ---------------------------------------------------------------------------
__global__ void gather_sigmoid_norm(const float* __restrict__ features,
                                    const int* __restrict__ idx,
                                    _Float16* __restrict__ out_h,
                                    float* __restrict__ out_norm,
                                    int rows) {
  int warp = threadIdx.x >> 5;
  int lane = threadIdx.x & 31;
  int row  = blockIdx.x * 8 + warp;
  if (row >= rows) return;
  int src = idx[row];
  const float* f = features + (size_t)src * kC;
  _Float16* oh = out_h + (size_t)row * kC;
  float nrm = 0.f;
  #pragma unroll
  for (int c = lane; c < kC; c += 32) {
    float s = sigmoidf_fast(f[c]);
    oh[c] = (_Float16)s;
    nrm += s * s;
  }
  #pragma unroll
  for (int off = 16; off >= 1; off >>= 1)
    nrm += __shfl_xor(nrm, off, 32);
  if (lane == 0) out_norm[row] = nrm;
}

// ---------------------------------------------------------------------------
// Kernel B: hard mining. 4 waves per block, each wave owns one 16-anchor
// strip; B-tiles staged once per block into double-buffered LDS via async
// global->LDS DMA, consumed by v_wmma_f32_16x16x32_f16 from ds_load_b128.
// ---------------------------------------------------------------------------
__global__ void __launch_bounds__(128)
hard_mine_wmma(const _Float16* __restrict__ pos_h,
               const _Float16* __restrict__ neg_h,
               const float* __restrict__ pos_nrm,
               const float* __restrict__ neg_nrm,
               int* __restrict__ hp, int* __restrict__ hn) {
  __shared__ __align__(16) char lds_raw[2 * kLdsTileB];

  const int blocks_per_class = kP / 64;   // 32 (4 strips of 16 per block)
  int k    = blockIdx.x / blocks_per_class;
  int sb   = blockIdx.x % blocks_per_class;
  int wave = threadIdx.x >> 5;
  int lane = threadIdx.x & 31;
  int tid  = threadIdx.x;
  int row0 = sb * 64 + wave * 16;
  int r16  = lane & 15;
  int hi   = lane >> 4;

  const _Float16* posk = pos_h + (size_t)k * kP * kC;
  const _Float16* negk = neg_h + (size_t)k * kQ * kC;
  const float* xxk = pos_nrm + (size_t)k * kP;
  const float* yyk = neg_nrm + (size_t)k * kQ;

  // A fragments for this wave's 16x256 anchor strip stay resident (64 VGPRs).
  v16h afrag[8];
  #pragma unroll
  for (int kk = 0; kk < 8; ++kk) afrag[kk] = load_frag_global(posk, row0, kk, lane);

  // Per-lane anchor-row norms: C/D layout has M = r + 8*hi for acc register r.
  float xs[8];
  #pragma unroll
  for (int r = 0; r < 8; ++r) xs[r] = xxk[row0 + r + 8 * hi];

  float bmax[8]; int imax[8];
  float bmin[8]; int imin[8];
  #pragma unroll
  for (int r = 0; r < 8; ++r) {
    bmax[r] = -1.0f;   imax[r] = 0;
    bmin[r] = 3.4e38f; imin[r] = 0;
  }

  // ---- phase 1: argmax over pos-pos squared distances ----
  stage_tile(posk, lds_raw, tid);
  wait_stage();
  for (int jt = 0; jt < kP / 16; ++jt) {
    const char* cur = lds_raw + (size_t)(jt & 1) * kLdsTileB;
    if (jt + 1 < kP / 16)
      stage_tile(posk + (size_t)(jt + 1) * 16 * kC,
                 lds_raw + (size_t)((jt + 1) & 1) * kLdsTileB, tid);
    v8f acc = {};
    #pragma unroll
    for (int kk = 0; kk < 8; ++kk) {
      v16h b = load_frag_lds(cur, kk, lane);
      acc = __builtin_amdgcn_wmma_f32_16x16x32_f16(false, afrag[kk], false, b,
                                                   (short)0, acc, false, false);
    }
    int col  = jt * 16 + r16;          // N = lane & 15
    float yv = xxk[col];
    #pragma unroll
    for (int r = 0; r < 8; ++r) {
      float d = fmaxf(xs[r] + yv - 2.0f * acc[r], 0.0f);
      if (d > bmax[r]) { bmax[r] = d; imax[r] = col; }
    }
    wait_stage();
  }

  // ---- phase 2: argmin over pos-neg squared distances ----
  stage_tile(negk, lds_raw, tid);
  wait_stage();
  for (int jt = 0; jt < kQ / 16; ++jt) {
    const char* cur = lds_raw + (size_t)(jt & 1) * kLdsTileB;
    if (jt + 1 < kQ / 16)
      stage_tile(negk + (size_t)(jt + 1) * 16 * kC,
                 lds_raw + (size_t)((jt + 1) & 1) * kLdsTileB, tid);
    v8f acc = {};
    #pragma unroll
    for (int kk = 0; kk < 8; ++kk) {
      v16h b = load_frag_lds(cur, kk, lane);
      acc = __builtin_amdgcn_wmma_f32_16x16x32_f16(false, afrag[kk], false, b,
                                                   (short)0, acc, false, false);
    }
    int col  = jt * 16 + r16;
    float yv = yyk[col];
    #pragma unroll
    for (int r = 0; r < 8; ++r) {
      float d = fmaxf(xs[r] + yv - 2.0f * acc[r], 0.0f);
      if (d < bmin[r]) { bmin[r] = d; imin[r] = col; }
    }
    wait_stage();
  }

  // Cross-lane tournament within each 16-lane group (ties -> lowest index,
  // matching jnp.argmax/argmin first-occurrence semantics).
  #pragma unroll
  for (int off = 8; off >= 1; off >>= 1) {
    #pragma unroll
    for (int r = 0; r < 8; ++r) {
      float ov = __shfl_xor(bmax[r], off, 32);
      int   oi = __shfl_xor(imax[r], off, 32);
      if (ov > bmax[r] || (ov == bmax[r] && oi < imax[r])) { bmax[r] = ov; imax[r] = oi; }
      float mv = __shfl_xor(bmin[r], off, 32);
      int   mi = __shfl_xor(imin[r], off, 32);
      if (mv < bmin[r] || (mv == bmin[r] && mi < imin[r])) { bmin[r] = mv; imin[r] = mi; }
    }
  }

  int* hpk = hp + (size_t)k * kP;
  int* hnk = hn + (size_t)k * kP;
  if (r16 == 0) {  // lanes 0 (rows 0-7) and 16 (rows 8-15) of each wave
    #pragma unroll
    for (int r = 0; r < 8; ++r) {
      hpk[row0 + r + 8 * hi] = imax[r];
      hnk[row0 + r + 8 * hi] = imin[r];
    }
  }
}

// ---------------------------------------------------------------------------
// Kernel C: exact fp32 pdist to mined indices + hinge; atomic scalar reduce.
// One wave32 per anchor.
// ---------------------------------------------------------------------------
__global__ void triplet_loss(const float* __restrict__ features,
                             const int* __restrict__ pos_idx,
                             const int* __restrict__ neg_idx,
                             const int* __restrict__ hp,
                             const int* __restrict__ hn,
                             float* __restrict__ out) {
  int warp = threadIdx.x >> 5;
  int lane = threadIdx.x & 31;
  int g = blockIdx.x * 8 + warp;          // anchor id over K*P
  if (g >= kK * kP) return;
  int k = g / kP;
  int p = g % kP;
  int srcA = pos_idx[(size_t)k * kP + p];
  int srcP = pos_idx[(size_t)k * kP + hp[g]];
  int srcN = neg_idx[(size_t)k * kQ + hn[g]];
  const float* fa = features + (size_t)srcA * kC;
  const float* fp = features + (size_t)srcP * kC;
  const float* fn = features + (size_t)srcN * kC;
  float sp = 0.f, sn = 0.f;
  #pragma unroll
  for (int c = lane; c < kC; c += 32) {
    float a  = sigmoidf_fast(fa[c]);
    float bp = sigmoidf_fast(fp[c]);
    float bn = sigmoidf_fast(fn[c]);
    float dp = a - bp, dn = a - bn;
    sp += dp * dp;
    sn += dn * dn;
  }
  #pragma unroll
  for (int off = 16; off >= 1; off >>= 1) {
    sp += __shfl_xor(sp, off, 32);
    sn += __shfl_xor(sn, off, 32);
  }
  if (lane == 0) {
    float d_p = sqrtf(sp + kEps);
    float d_n = sqrtf(sn + kEps);
    float contrib = fmaxf(kMargin + d_p - d_n, 0.0f) / (float)kP;
    atomicAdd(out, contrib);
  }
}

extern "C" void kernel_launch(void* const* d_in, const int* in_sizes, int n_in,
                              void* d_out, int out_size, void* d_ws, size_t ws_size,
                              hipStream_t stream) {
  (void)in_sizes; (void)n_in; (void)out_size; (void)ws_size;
  const float* features = (const float*)d_in[0];
  const int* pos_idx = (const int*)d_in[1];
  const int* neg_idx = (const int*)d_in[2];
  float* out = (float*)d_out;

  // Workspace layout (~42.6 MB, fully L2-resident on MI455X):
  char* ws = (char*)d_ws;
  _Float16* pos_h = (_Float16*)ws;  ws += (size_t)kK * kP * kC * sizeof(_Float16);
  _Float16* neg_h = (_Float16*)ws;  ws += (size_t)kK * kQ * kC * sizeof(_Float16);
  float* pos_nrm = (float*)ws;      ws += (size_t)kK * kP * sizeof(float);
  float* neg_nrm = (float*)ws;      ws += (size_t)kK * kQ * sizeof(float);
  int* hp = (int*)ws;               ws += (size_t)kK * kP * sizeof(int);
  int* hn = (int*)ws;

  const int prows = kK * kP;  // 40960
  const int nrows = kK * kQ;  // 40960
  gather_sigmoid_norm<<<prows / 8, 256, 0, stream>>>(features, pos_idx, pos_h, pos_nrm, prows);
  gather_sigmoid_norm<<<nrows / 8, 256, 0, stream>>>(features, neg_idx, neg_h, neg_nrm, nrows);

  hard_mine_wmma<<<kK * (kP / 64), 128, 0, stream>>>(pos_h, neg_h, pos_nrm, neg_nrm, hp, hn);

  (void)hipMemsetAsync(d_out, 0, sizeof(float), stream);
  triplet_loss<<<(kK * kP) / 8, 256, 0, stream>>>(features, pos_idx, neg_idx, hp, hn, out);
}
